// DANetHead_19481971655244
// MI455X (gfx1250) — matrix operand order repaired and checked by reading, not compiled, not run on hardware
//
#include <hip/hip_runtime.h>
#include <hip/hip_bf16.h>

// ---------------------------------------------------------------------------
// CDNA5 (gfx1250) DANet head.
//  - All matmuls via V_WMMA_F32_16X16X32_BF16 (bf16 A/B, fp32 accum).
//  - Activations kept K-major ([N, C]) so every GEMM B tile is a straight 2D
//    copy -> staged into LDS with the Tensor Data Mover (tensor_load_to_lds,
//    TENSORcnt), consumed with ds_load_b128 fragment reads.
//  - PAM is flash-attention style: the 4096x4096 energy matrix is never
//    materialized (would be 256 MB of HBM traffic at 23.3 TB/s for nothing).
// ---------------------------------------------------------------------------

typedef __attribute__((ext_vector_type(16))) __bf16 v16bf;
typedef __attribute__((ext_vector_type(8)))  float  v8f;
typedef __attribute__((ext_vector_type(4)))  float  f32x4;
typedef __attribute__((ext_vector_type(4)))  unsigned int u32x4;
typedef __attribute__((ext_vector_type(8)))  int    i32x8;
typedef __attribute__((ext_vector_type(4)))  int    i32x4;

union FragBF  { v16bf v; f32x4 f4[2]; };
union Pack8BF { f32x4 q; __bf16 h[8]; };

__device__ __forceinline__ v8f wmma_bf16(v16bf a, v16bf b, v8f c) {
  return __builtin_amdgcn_wmma_f32_16x16x32_bf16(
      /*neg_a=*/false, a, /*neg_b=*/false, b,
      /*c_mod=*/(short)0, c, /*reuse_a=*/false, /*reuse_b=*/false);
}

// A-matrix 16x32 bf16 fragment. lane<16: row M=lane, K=0..15;
// lane>=16: row M=lane-16, K=16..31.
__device__ __forceinline__ v16bf load_a_frag(const __bf16* base, int ld) {
  int lane = threadIdx.x & 31;
  const __bf16* p = base + (size_t)(lane & 15) * ld + ((lane & 16) ? 16 : 0);
  FragBF fr;
  fr.f4[0] = *reinterpret_cast<const f32x4*>(p);
  fr.f4[1] = *reinterpret_cast<const f32x4*>(p + 8);
  return fr.v;
}

// B-matrix 32x16 fragment: p -> this lane's 16 contiguous K values.
__device__ __forceinline__ v16bf load_b_frag(const __bf16* p) {
  FragBF fr;
  fr.f4[0] = *reinterpret_cast<const f32x4*>(p);
  fr.f4[1] = *reinterpret_cast<const f32x4*>(p + 8);
  return fr.v;
}

// ---------------------------------------------------------------------------
// TDM: async 2D tile load Global -> LDS (bf16 elements).
// D# per CDNA5 ISA ch.8: group0 = {count/lds_addr/global_addr/type},
// group1 = {data_size, tensor dims, tile dims, strides}. 2D: groups 2/3 null.
// ---------------------------------------------------------------------------
__device__ __forceinline__ void tdm_load_2d_bf16(unsigned int ldsAddr,
                                                 const void* gaddr,
                                                 unsigned int rowLen,     // tensor_dim0 (elems)
                                                 unsigned int numRows,    // tensor_dim1
                                                 unsigned int rowStride,  // tensor_dim0_stride (elems)
                                                 unsigned int tileK,      // tile_dim0
                                                 unsigned int tileN) {    // tile_dim1
  unsigned long long ga = (unsigned long long)(size_t)gaddr;
  u32x4 g0;
  g0.x = 1u;                                   // count=1, user descriptor
  g0.y = ldsAddr;                              // lds_addr (bytes)
  g0.z = (unsigned int)ga;                     // global_addr[31:0]
  g0.w = (unsigned int)((ga >> 32) & 0x01FFFFFFu) | (2u << 30);  // addr[56:32] | type=2
  i32x8 g1;
  g1[0] = (int)(1u << 16);                                     // data_size = 2 bytes
  g1[1] = (int)((rowLen & 0xFFFFu) << 16);                     // tensor_dim0[15:0]
  g1[2] = (int)(((rowLen >> 16) & 0xFFFFu) | ((numRows & 0xFFFFu) << 16));
  g1[3] = (int)(((numRows >> 16) & 0xFFFFu) | ((tileK & 0xFFFFu) << 16));  // tile_dim0
  g1[4] = (int)(tileN & 0xFFFFu);                              // tile_dim1 (tile_dim2=0)
  g1[5] = (int)rowStride;                                      // dim0_stride[31:0]
  g1[6] = 0;                                                   // dim0_stride[47:32]
  g1[7] = 0;
  i32x4 z4 = {0, 0, 0, 0};
#if defined(__clang_major__) && __clang_major__ >= 23
  i32x8 z8 = {0, 0, 0, 0, 0, 0, 0, 0};
  __builtin_amdgcn_tensor_load_to_lds(g0, g1, z4, z4, z8, 0);
#else
  __builtin_amdgcn_tensor_load_to_lds(g0, g1, z4, z4, 0);
#endif
}

// ---------------------------------------------------------------------------
// fp32 -> bf16 convert (weights)
// ---------------------------------------------------------------------------
__global__ void cvt_f32_to_bf16(const float* __restrict__ in,
                                __bf16* __restrict__ out, int n) {
  int i = blockIdx.x * blockDim.x + threadIdx.x;
  if (i < n) out[i] = (__bf16)in[i];
}

// ---------------------------------------------------------------------------
// x [B, C, N] fp32 -> xT [B, N, C] bf16 (32x32 LDS tile transpose)
// ---------------------------------------------------------------------------
__global__ __launch_bounds__(256)
void transpose_cvt(const float* __restrict__ in, __bf16* __restrict__ outp,
                   int C, int Nn) {
  __shared__ float tile[32][33];
  const int bz = blockIdx.z;
  const float* src = in  + (size_t)bz * C * Nn;
  __bf16*      dst = outp + (size_t)bz * C * Nn;
  const int n0 = blockIdx.x * 32, c0 = blockIdx.y * 32;
#pragma unroll
  for (int i = 0; i < 4; ++i)
    tile[threadIdx.y + i * 8][threadIdx.x] =
        src[(size_t)(c0 + threadIdx.y + i * 8) * Nn + n0 + threadIdx.x];
  __syncthreads();
#pragma unroll
  for (int i = 0; i < 4; ++i)
    dst[(size_t)(n0 + threadIdx.y + i * 8) * C + c0 + threadIdx.x] =
        (__bf16)tile[threadIdx.x][threadIdx.y + i * 8];
}

// ---------------------------------------------------------------------------
// Generic WMMA GEMM: C[M,N] = A[M,K] @ B, B K-major: element(k,n)=Bm[n*ldb+k].
// Block = 4 waves, 64(M) x 64(N) tile; B tile via TDM -> LDS; A direct global.
// Epilogue: gamma*(A@B) + bias + residB(bf16) + residF(f32), optional relu;
// stores: normal bf16 / normal f32 / transposed bf16 (vectorized).
// ---------------------------------------------------------------------------
__global__ __launch_bounds__(128)
void gemm_bf16_wmma(const __bf16* __restrict__ A, int lda, long aBatch,
                    const __bf16* __restrict__ Bm, int ldb, long bBatch,
                    const float* __restrict__ bias,
                    const __bf16* __restrict__ residB, long residBBatch,
                    const float*  __restrict__ residF, long residFBatch,
                    const float*  __restrict__ gammaPtr, int relu,
                    __bf16* __restrict__ Cbf, float* __restrict__ Cf,
                    int ldc, long cBatch,
                    __bf16* __restrict__ CT, int ldct, long ctBatch,
                    int M, int N, int K) {
  __shared__ __attribute__((aligned(16))) __bf16 Bs[64 * 32];   // [n][k]

  const int tid  = threadIdx.x;
  const int wave = tid >> 5;
  const int lane = tid & 31;
  const int bz   = blockIdx.z;
  const int n0   = blockIdx.x * 64;
  const int m0   = blockIdx.y * 64;

  const __bf16* Ab = A  + (size_t)bz * aBatch;
  const __bf16* Bb = Bm + (size_t)bz * bBatch + (size_t)n0 * ldb;

  const bool active = (m0 + wave * 16) < M;   // wave-uniform
  const int col = lane & 15;
  const int kb  = (lane & 16) ? 16 : 0;
  const int arow = m0 + wave * 16 + col;
  const __bf16* Ap = Ab + (size_t)(active ? arow : 0) * lda + ((lane & 16) ? 16 : 0);

  const unsigned int ldsAddr = (unsigned int)(size_t)(void*)&Bs[0];

  v8f acc[4] = {};

  for (int kt = 0; kt < K; kt += 32) {
    if (wave == 0) {
      tdm_load_2d_bf16(ldsAddr, Bb + kt, (unsigned)K, (unsigned)N,
                       (unsigned)ldb, 32u, 64u);
      __builtin_amdgcn_s_wait_tensorcnt((short)0);
    }
    __syncthreads();
    if (active) {
      if (kt + 32 < K) __builtin_prefetch(Ap + kt + 32, 0, 1);
      FragBF af;
      af.f4[0] = *reinterpret_cast<const f32x4*>(Ap + kt);
      af.f4[1] = *reinterpret_cast<const f32x4*>(Ap + kt + 8);
#pragma unroll
      for (int j = 0; j < 4; ++j) {
        v16bf bf = load_b_frag(&Bs[(j * 16 + col) * 32 + kb]);
        acc[j] = wmma_bf16(af.v, bf, acc[j]);
      }
    }
    // drain this wave's LDS reads before TDM recycles the tile
    asm volatile("s_wait_dscnt 0" ::: "memory");
    __syncthreads();
  }

  if (!active) return;
  const float g = gammaPtr ? gammaPtr[0] : 1.0f;
  const int mbase = m0 + wave * 16 + 8 * (lane >> 4);
#pragma unroll
  for (int j = 0; j < 4; ++j) {
    const int gn = n0 + j * 16 + col;
    float vals[8];
#pragma unroll
    for (int r = 0; r < 8; ++r) {
      float v = acc[j][r] * g;
      int gm = mbase + r;
      if (bias)   v += bias[gm];
      if (residB) v += (float)residB[(size_t)bz * residBBatch + (size_t)gm * ldc + gn];
      if (residF) v += residF[(size_t)bz * residFBatch + (size_t)gm * ldc + gn];
      if (relu)   v = fmaxf(v, 0.0f);
      vals[r] = v;
    }
    if (Cbf) {
#pragma unroll
      for (int r = 0; r < 8; ++r)
        Cbf[(size_t)bz * cBatch + (size_t)(mbase + r) * ldc + gn] = (__bf16)vals[r];
    }
    if (Cf) {
#pragma unroll
      for (int r = 0; r < 8; ++r)
        Cf[(size_t)bz * cBatch + (size_t)(mbase + r) * ldc + gn] = vals[r];
    }
    if (CT) {   // transposed [n, m] store: 8 consecutive m -> one 16B store
      Pack8BF pk;
#pragma unroll
      for (int r = 0; r < 8; ++r) pk.h[r] = (__bf16)vals[r];
      *reinterpret_cast<f32x4*>(&CT[(size_t)bz * ctBatch + (size_t)gn * ldct + mbase]) = pk.q;
    }
  }
}

// ---------------------------------------------------------------------------
// PAM flash attention. qT/kT: [B, N, 32] bf16; V: [B, 128, N] bf16;
// feat1T/saT: [B, N, 128] bf16. Per wave: 16 queries, online softmax over all
// keys in 32-key steps; saT = gamma * out + feat1.
// ---------------------------------------------------------------------------
__global__ __launch_bounds__(128)
void pam_flash(const __bf16* __restrict__ qT, const __bf16* __restrict__ kT,
               const __bf16* __restrict__ Vb, const __bf16* __restrict__ f1T,
               const float* __restrict__ gammaPtr,
               __bf16* __restrict__ saT, int Nseq, int Cv) {
  __shared__ __attribute__((aligned(16))) __bf16 Ps[4][16 * 32];

  const int tid  = threadIdx.x;
  const int wave = tid >> 5;
  const int lane = tid & 31;
  const int bz   = blockIdx.z;
  const int nq0  = blockIdx.x * 64 + wave * 16;

  const size_t qoff = (size_t)bz * Nseq * 32;
  const size_t boff = (size_t)bz * Cv * Nseq;   // batch offset for V / f1T / saT

  const v16bf qf = load_a_frag(qT + qoff + (size_t)nq0 * 32, 32);

  v8f   oacc[8] = {};
  float mrow[8], lrow[8];
#pragma unroll
  for (int r = 0; r < 8; ++r) { mrow[r] = -3.0e38f; lrow[r] = 0.0f; }

  const int col = lane & 15;
  const int kb  = (lane & 16) ? 16 : 0;

  for (int mt = 0; mt < Nseq; mt += 32) {
    v16bf k0 = load_b_frag(kT + qoff + (size_t)(mt + col) * 32 + kb);
    v16bf k1 = load_b_frag(kT + qoff + (size_t)(mt + 16 + col) * 32 + kb);
    v8f z = {};
    v8f e0 = wmma_bf16(qf, k0, z);
    v8f e1 = wmma_bf16(qf, k1, z);

#pragma unroll
    for (int r = 0; r < 8; ++r) {
      float a = e0[r], c = e1[r];
      float t = fmaxf(a, c);
#pragma unroll
      for (int off = 8; off >= 1; off >>= 1) t = fmaxf(t, __shfl_xor(t, off, 32));
      float newm  = fmaxf(mrow[r], t);
      float alpha = __expf(mrow[r] - newm);
      float p0    = __expf(a - newm);
      float p1    = __expf(c - newm);
      float s     = p0 + p1;
#pragma unroll
      for (int off = 8; off >= 1; off >>= 1) s += __shfl_xor(s, off, 32);
      lrow[r] = lrow[r] * alpha + s;
      mrow[r] = newm;
#pragma unroll
      for (int j = 0; j < 8; ++j) oacc[j][r] = oacc[j][r] * alpha;
      int rowIdx = r + 8 * (lane >> 4);
      Ps[wave][rowIdx * 32 + col]      = (__bf16)p0;
      Ps[wave][rowIdx * 32 + 16 + col] = (__bf16)p1;
    }
    __syncthreads();

    v16bf pf = load_a_frag(&Ps[wave][0], 32);
#pragma unroll
    for (int j = 0; j < 8; ++j) {
      v16bf vf = load_b_frag(Vb + boff + (size_t)(j * 16 + col) * Nseq + mt + kb);
      oacc[j] = wmma_bf16(pf, vf, oacc[j]);
    }
    __syncthreads();
  }

  const float g = gammaPtr[0];
#pragma unroll
  for (int j = 0; j < 8; ++j) {
#pragma unroll
    for (int r = 0; r < 8; ++r) {
      int n = nq0 + r + 8 * (lane >> 4);
      int c = j * 16 + col;
      float o   = oacc[j][r] / lrow[r];
      float val = g * o + (float)f1T[boff + (size_t)n * Cv + c];
      saT[boff + (size_t)n * Cv + c] = (__bf16)val;
    }
  }
}

// ---------------------------------------------------------------------------
// CAM softmax: attn = softmax(rowmax(E) - E) = exp(rowmin - e) / sum
// ---------------------------------------------------------------------------
__global__ __launch_bounds__(128)
void cam_softmax(const float* __restrict__ energy, __bf16* __restrict__ attn, int C) {
  __shared__ float red[128];
  const int row = blockIdx.x, bz = blockIdx.z, t = threadIdx.x;
  const size_t off = ((size_t)bz * C + row) * C;
  float e = energy[off + t];

  red[t] = e; __syncthreads();
  for (int s = 64; s > 0; s >>= 1) {
    if (t < s) red[t] = fminf(red[t], red[t + s]);
    __syncthreads();
  }
  float mn = red[0]; __syncthreads();

  float p = __expf(mn - e);
  red[t] = p; __syncthreads();
  for (int s = 64; s > 0; s >>= 1) {
    if (t < s) red[t] += red[t + s];
    __syncthreads();
  }
  attn[off + t] = (__bf16)(p / red[0]);
}

// ---------------------------------------------------------------------------
// Host orchestration
// ---------------------------------------------------------------------------
extern "C" void kernel_launch(void* const* d_in, const int* in_sizes, int n_in,
                              void* d_out, int out_size, void* d_ws, size_t ws_size,
                              hipStream_t stream) {
  (void)in_sizes; (void)n_in; (void)out_size; (void)ws_size;

  constexpr int  Bb_ = 4, Cin_ = 512, Ns_ = 4096, Ci_ = 128, Cq_ = 32, Co_ = 256;
  constexpr long CiN = (long)Ci_ * Ns_;
  constexpr long CoN = (long)Co_ * Ns_;

  const float* x    = (const float*)d_in[0];
  const float* Wa   = (const float*)d_in[1];
  const float* Wc   = (const float*)d_in[2];
  const float* Wq   = (const float*)d_in[3];
  const float* bq   = (const float*)d_in[4];
  const float* Wk   = (const float*)d_in[5];
  const float* bk   = (const float*)d_in[6];
  const float* Wv   = (const float*)d_in[7];
  const float* bv   = (const float*)d_in[8];
  const float* gpam = (const float*)d_in[9];
  const float* gcam = (const float*)d_in[10];
  const float* Wa1  = (const float*)d_in[11];
  const float* Wc1  = (const float*)d_in[12];
  const float* W1   = (const float*)d_in[13];
  const float* b1   = (const float*)d_in[14];
  const float* W2   = (const float*)d_in[15];
  const float* b2   = (const float*)d_in[16];
  const float* W3   = (const float*)d_in[17];
  const float* b3   = (const float*)d_in[18];
  float* out = (float*)d_out;

  char* base = (char*)d_ws;
  size_t off = 0;
  auto alloc_bf = [&](size_t elems) {
    __bf16* p = (__bf16*)(base + off);
    off = (off + elems * 2 + 255) & ~(size_t)255;
    return p;
  };
  auto alloc_f = [&](size_t elems) {
    float* p = (float*)(base + off);
    off = (off + elems * 4 + 255) & ~(size_t)255;
    return p;
  };

  __bf16* xT     = alloc_bf((size_t)Bb_ * Cin_ * Ns_);   // [B, N, 512]
  __bf16* Wabf   = alloc_bf((size_t)Ci_ * Cin_);
  __bf16* Wcbf   = alloc_bf((size_t)Ci_ * Cin_);
  __bf16* Wqbf   = alloc_bf((size_t)Cq_ * Ci_);
  __bf16* Wkbf   = alloc_bf((size_t)Cq_ * Ci_);
  __bf16* Wvbf   = alloc_bf((size_t)Ci_ * Ci_);
  __bf16* Wa1bf  = alloc_bf((size_t)Ci_ * Ci_);
  __bf16* Wc1bf  = alloc_bf((size_t)Ci_ * Ci_);
  __bf16* W1bf   = alloc_bf((size_t)Co_ * Ci_);
  __bf16* W2bf   = alloc_bf((size_t)Co_ * Ci_);
  __bf16* W3bf   = alloc_bf((size_t)Co_ * Co_);
  __bf16* feat1T = alloc_bf((size_t)Bb_ * CiN);          // [B, N, 128]
  __bf16* feat2  = alloc_bf((size_t)Bb_ * CiN);          // [B, 128, N]
  __bf16* feat2T = alloc_bf((size_t)Bb_ * CiN);          // [B, N, 128]
  __bf16* qTb    = alloc_bf((size_t)Bb_ * Ns_ * Cq_);    // [B, N, 32]
  __bf16* kTb    = alloc_bf((size_t)Bb_ * Ns_ * Cq_);
  __bf16* vB     = alloc_bf((size_t)Bb_ * CiN);          // [B, 128, N]
  __bf16* saT    = alloc_bf((size_t)Bb_ * CiN);          // [B, N, 128]
  __bf16* sa2T   = alloc_bf((size_t)Bb_ * CiN);
  float*  saOut  = alloc_f((size_t)Bb_ * CoN);           // [B, 256, N] fp32
  float*  energ  = alloc_f((size_t)Bb_ * Ci_ * Ci_);
  __bf16* attnB  = alloc_bf((size_t)Bb_ * Ci_ * Ci_);
  __bf16* scT    = alloc_bf((size_t)Bb_ * CiN);
  __bf16* sc2T   = alloc_bf((size_t)Bb_ * CiN);
  __bf16* sSumT  = alloc_bf((size_t)Bb_ * CoN);          // [B, N, 256]

  auto cvt = [&](const float* in, __bf16* o, int n) {
    cvt_f32_to_bf16<<<dim3((n + 255) / 256), dim3(256), 0, stream>>>(in, o, n);
  };
  transpose_cvt<<<dim3(Ns_ / 32, Cin_ / 32, Bb_), dim3(32, 8), 0, stream>>>(
      x, xT, Cin_, Ns_);
  cvt(Wa,  Wabf,  Ci_ * Cin_);   cvt(Wc,  Wcbf,  Ci_ * Cin_);
  cvt(Wq,  Wqbf,  Cq_ * Ci_);    cvt(Wk,  Wkbf,  Cq_ * Ci_);
  cvt(Wv,  Wvbf,  Ci_ * Ci_);
  cvt(Wa1, Wa1bf, Ci_ * Ci_);    cvt(Wc1, Wc1bf, Ci_ * Ci_);
  cvt(W1,  W1bf,  Co_ * Ci_);    cvt(W2,  W2bf,  Co_ * Ci_);
  cvt(W3,  W3bf,  Co_ * Co_);

  auto gemm = [&](const __bf16* A, int lda, long aB,
                  const __bf16* Bm, int ldb, long bB,
                  const float* bias, const __bf16* rB, long rBB,
                  const float* rF, long rFB, const float* gam, int relu,
                  __bf16* Cb, float* Cf2, int ldc, long cB,
                  __bf16* CT, int ldct, long ctB, int M, int N, int K) {
    dim3 g(N / 64, (M + 63) / 64, Bb_);
    gemm_bf16_wmma<<<g, dim3(128), 0, stream>>>(
        A, lda, aB, Bm, ldb, bB, bias, rB, rBB, rF, rFB, gam, relu,
        Cb, Cf2, ldc, cB, CT, ldct, ctB, M, N, K);
  };

  // feat1T = relu(Wa @ x)^T ; feat2 (+T) = relu(Wc @ x)
  gemm(Wabf, Cin_, 0, xT, Cin_, (long)Cin_ * Ns_, nullptr, nullptr, 0,
       nullptr, 0, nullptr, 1, nullptr, nullptr, Ns_, 0,
       feat1T, Ci_, CiN, Ci_, Ns_, Cin_);
  gemm(Wcbf, Cin_, 0, xT, Cin_, (long)Cin_ * Ns_, nullptr, nullptr, 0,
       nullptr, 0, nullptr, 1, feat2, nullptr, Ns_, CiN,
       feat2T, Ci_, CiN, Ci_, Ns_, Cin_);

  // q^T, k^T ([N,32]) and v ([128,N])
  gemm(Wqbf, Ci_, 0, feat1T, Ci_, CiN, bq, nullptr, 0, nullptr, 0, nullptr, 0,
       nullptr, nullptr, Ns_, 0, qTb, Cq_, (long)Ns_ * Cq_, Cq_, Ns_, Ci_);
  gemm(Wkbf, Ci_, 0, feat1T, Ci_, CiN, bk, nullptr, 0, nullptr, 0, nullptr, 0,
       nullptr, nullptr, Ns_, 0, kTb, Cq_, (long)Ns_ * Cq_, Cq_, Ns_, Ci_);
  gemm(Wvbf, Ci_, 0, feat1T, Ci_, CiN, bv, nullptr, 0, nullptr, 0, nullptr, 0,
       vB, nullptr, Ns_, CiN, nullptr, 0, 0, Ci_, Ns_, Ci_);

  // PAM flash attention -> saT = gamma*out + feat1
  pam_flash<<<dim3(Ns_ / 64, 1, Bb_), dim3(128), 0, stream>>>(
      qTb, kTb, vB, feat1T, gpam, saT, Ns_, Ci_);

  // sa2T = relu(Wa1 @ sa)^T ; saOut = W1 @ sa2 + b1 (fp32, [256,N])
  gemm(Wa1bf, Ci_, 0, saT, Ci_, CiN, nullptr, nullptr, 0, nullptr, 0,
       nullptr, 1, nullptr, nullptr, Ns_, 0, sa2T, Ci_, CiN, Ci_, Ns_, Ci_);
  gemm(W1bf, Ci_, 0, sa2T, Ci_, CiN, b1, nullptr, 0, nullptr, 0,
       nullptr, 0, nullptr, saOut, Ns_, CoN, nullptr, 0, 0, Co_, Ns_, Ci_);

  // CAM: energy = feat2 @ feat2^T (both operands natural [c,n] layout)
  gemm(feat2, Ns_, CiN, feat2, Ns_, CiN, nullptr, nullptr, 0, nullptr, 0,
       nullptr, 0, nullptr, energ, Ci_, (long)Ci_ * Ci_, nullptr, 0, 0,
       Ci_, Ci_, Ns_);
  cam_softmax<<<dim3(Ci_, 1, Bb_), dim3(128), 0, stream>>>(energ, attnB, Ci_);
  // scT = (gamma * attn @ feat2 + feat2)^T
  gemm(attnB, Ci_, (long)Ci_ * Ci_, feat2T, Ci_, CiN, nullptr,
       feat2, CiN, nullptr, 0, gcam, 0, nullptr, nullptr, Ns_, 0,
       scT, Ci_, CiN, Ci_, Ns_, Ci_);

  // sc2T = relu(Wc1 @ sc)^T ; sSumT = (W2 @ sc2 + b2 + saOut)^T
  gemm(Wc1bf, Ci_, 0, scT, Ci_, CiN, nullptr, nullptr, 0, nullptr, 0,
       nullptr, 1, nullptr, nullptr, Ns_, 0, sc2T, Ci_, CiN, Ci_, Ns_, Ci_);
  gemm(W2bf, Ci_, 0, sc2T, Ci_, CiN, b2, nullptr, 0, saOut, CoN,
       nullptr, 0, nullptr, nullptr, Ns_, 0, sSumT, Co_, CoN, Co_, Ns_, Ci_);

  // final: out = W3 @ s_sum + b3 (fp32 -> d_out)
  gemm(W3bf, Co_, 0, sSumT, Co_, CoN, b3, nullptr, 0, nullptr, 0,
       nullptr, 0, nullptr, out, Ns_, CoN, nullptr, 0, 0, Co_, Ns_, Co_);
}
